// TransformerLayer_87943750353244
// MI455X (gfx1250) — matrix operand order repaired and checked by reading, not compiled, not run on hardware
//
#include <hip/hip_runtime.h>
#include <hip/hip_bf16.h>

typedef __attribute__((ext_vector_type(16))) _Float16 v16h;
typedef __attribute__((ext_vector_type(8)))  float    v8f;

#define B_  2
#define S_  4096
#define E_  512
#define H_  8
#define D_  64
#define W_  512
#define NW_ 8
#define HD_ 512   // H*D

union F16x16 { v16h v; uint4 q[2]; };

static __device__ __forceinline__ v8f wmma16(v16h a, v16h b, v8f c) {
  // D = A(16x32 f16) x B(32x16 f16) + C(16x16 f32)
  return __builtin_amdgcn_wmma_f32_16x16x32_f16(false, a, false, b, (short)0, c, false, false);
}

// ---------------------------------------------------------------------------
// Kernel 0: fp32 -> fp16 conversions / layout transforms.
//   xs  (B,S,E)            -> xs_h (B,S,E) f16
//   w_{q,k,v} (E,H,D)      -> wt[m] (HD,E) f16  (transposed: contiguous-K B frags)
//   prev_k (B,W,H,D)       -> kbuf[b][0]  (slot 0 of (B,NW+1,W,H,D))
//   prev_v (B,W,H,D)       -> vT[b][0]    (slot 0 of (B,NW+1,H,D,W), j-contiguous)
// ---------------------------------------------------------------------------
__global__ void convert_kernel(const float* __restrict__ xs,
                               const float* __restrict__ wq,
                               const float* __restrict__ wk,
                               const float* __restrict__ wv,
                               const float* __restrict__ pk,
                               const float* __restrict__ pv,
                               _Float16* __restrict__ xs_h,
                               _Float16* __restrict__ wt,
                               _Float16* __restrict__ kbuf,
                               _Float16* __restrict__ vT) {
  long i = (long)blockIdx.x * blockDim.x + threadIdx.x;
  const long XSN = (long)B_ * S_ * E_;
  const long WTN = (long)E_ * HD_;
  const long PVN = (long)B_ * W_ * HD_;
  if (i < XSN) { xs_h[i] = (_Float16)xs[i]; return; }
  i -= XSN;
  if (i < 3 * WTN) {
    int  m = (int)(i / WTN);
    long t = i % WTN;
    int  n = (int)(t / E_);
    int  k = (int)(t % E_);
    const float* w = (m == 0) ? wq : (m == 1) ? wk : wv;
    wt[(long)m * WTN + t] = (_Float16)w[(long)k * HD_ + n];
    return;
  }
  i -= 3 * WTN;
  if (i < PVN) {
    int  b = (int)(i / ((long)W_ * HD_));
    long r = i % ((long)W_ * HD_);
    kbuf[(long)b * (NW_ + 1) * W_ * HD_ + r] = (_Float16)pk[i];
    return;
  }
  i -= PVN;
  if (i < PVN) {
    int  b   = (int)(i / ((long)W_ * HD_));
    long rem = i % ((long)W_ * HD_);
    int  wp  = (int)(rem / HD_);
    int  c   = (int)(rem % HD_);
    int  h   = c >> 6, d = c & 63;
    vT[(((long)b * (NW_ + 1)) * H_ + h) * (long)(D_ * W_) + (long)d * W_ + wp] =
        (_Float16)pv[i];
  }
}

// ---------------------------------------------------------------------------
// Kernel 1: QKV projection GEMM.  grid=(M/16, 3 mats), block=256 (8 waves).
// Each wave: one 16x64 output strip, K-loop over E=512 in WMMA K=32 steps.
// ---------------------------------------------------------------------------
__global__ __launch_bounds__(256) void proj_kernel(
    const _Float16* __restrict__ xs_h, const _Float16* __restrict__ wt,
    _Float16* __restrict__ qbuf, _Float16* __restrict__ kbuf,
    _Float16* __restrict__ vT,
    float* __restrict__ next_k, float* __restrict__ next_v) {
  const int mat  = blockIdx.y;
  const int row0 = blockIdx.x << 4;
  const int wave = threadIdx.x >> 5;
  const int lane = threadIdx.x & 31;
  const int h2 = lane >> 4, nl = lane & 15;
  const int col0 = wave * 64;

  const _Float16* wm   = wt + (long)mat * (E_ * HD_);
  const _Float16* arow = xs_h + (long)(row0 + nl) * E_;   // A: M = lane&15

  v8f acc[4];
  acc[0] = {}; acc[1] = {}; acc[2] = {}; acc[3] = {};

  for (int kk = 0; kk < E_; kk += 32) {
    // A frag: lanes 0-15 K=0..15, lanes 16-31 K=16..31 (contiguous, matches B)
    F16x16 a;
    a.q[0] = *(const uint4*)(arow + kk + 16 * h2);
    a.q[1] = *(const uint4*)(arow + kk + 16 * h2 + 8);
#pragma unroll
    for (int g = 0; g < 4; ++g) {
      // B frag: lane col n = nl, K = 16*h2..+15 contiguous in transposed weights
      const _Float16* bp = wm + (long)(col0 + 16 * g + nl) * E_ + kk + 16 * h2;
      F16x16 bf;
      bf.q[0] = *(const uint4*)(bp);
      bf.q[1] = *(const uint4*)(bp + 8);
      acc[g] = wmma16(a.v, bf.v, acc[g]);
    }
  }

#pragma unroll
  for (int g = 0; g < 4; ++g) {
    const int col = col0 + 16 * g + nl;
#pragma unroll
    for (int r = 0; r < 8; ++r) {
      const int row = row0 + 8 * h2 + r;           // C/D layout: M = 8*h2 + r
      const float val = acc[g][r];
      const int b = row >> 12, s = row & (S_ - 1);
      if (mat == 0) {
        qbuf[(long)row * HD_ + col] = (_Float16)(val * 0.125f);  // 1/sqrt(D)
      } else if (mat == 1) {
        kbuf[((long)(b * (NW_ + 1) + 1 + (s >> 9))) * (W_ * HD_) +
             (long)(s & (W_ - 1)) * HD_ + col] = (_Float16)val;
        if (s >= S_ - W_)
          next_k[((long)b * W_ + (s - (S_ - W_))) * HD_ + col] = val;
      } else {
        const int hh = col >> 6, d = col & 63;
        vT[(((long)(b * (NW_ + 1) + 1 + (s >> 9))) * H_ + hh) * (long)(D_ * W_) +
           (long)d * W_ + (s & (W_ - 1))] = (_Float16)val;
        if (s >= S_ - W_)
          next_v[((long)b * W_ + (s - (S_ - W_))) * HD_ + col] = val;
      }
    }
  }
}

// ---------------------------------------------------------------------------
// Kernel 2: windowed attention.  grid=(W/64, B*NW*H), block=128 (4 waves).
// Wave = 16 query rows.  Banded mask => only 33 of 64 column tiles needed.
// LDS: bias table 512 f32, per-wave 16x544 f32 scores + 16x544 f16 P.
// ---------------------------------------------------------------------------
#define SCW 544   // 34 tiles of 16 local columns (33 computed + 1 zero pad)

__global__ __launch_bounds__(128) void attn_kernel(
    const _Float16* __restrict__ qbuf, const _Float16* __restrict__ kbuf,
    const _Float16* __restrict__ vT, const float* __restrict__ rel_bias,
    float* __restrict__ out_attn) {
  extern __shared__ char smem[];
  float*    sc = (float*)smem;                              // 4*16*544 f32
  float*    tb = (float*)(smem + 4 * 16 * SCW * 4);         // 512 f32
  _Float16* Pb = (_Float16*)(smem + 4 * 16 * SCW * 4 + 512 * 4);  // 4*16*544 f16

  const int z = blockIdx.y;
  const int h = z & (H_ - 1);
  const int n = (z >> 3) & (NW_ - 1);
  const int b = z >> 6;
  const int wave = threadIdx.x >> 5;
  const int lane = threadIdx.x & 31;
  const int h2 = lane >> 4, nl = lane & 15;
  const int i0 = blockIdx.x * 64 + wave * 16;   // query row within window

  // T5 causal bucket bias table: tb[m] = rel_bias[h][bucket(m)], m = i+W-j
  for (int idx = (int)threadIdx.x; idx < 512; idx += 128) {
    int bucket;
    if (idx < 16) bucket = idx;
    else {
      int v = 16 + (int)(logf((float)idx * (1.0f / 16.0f)) * (16.0f / logf(8.0f)));
      bucket = v < 31 ? v : 31;
    }
    tb[idx] = rel_bias[h * 32 + bucket];
  }
  __syncthreads();

  float*    scw = sc + wave * 16 * SCW;
  _Float16* Pw  = Pb + wave * 16 * SCW;

  // pad tile (local cols 528..543) -> -inf so its P becomes exactly 0
  for (int t = lane; t < 256; t += 32)
    scw[(t >> 4) * SCW + 528 + (t & 15)] = -1e30f;

  // Q fragments: reused for all 33 score tiles. A row M = lane&15.
  const _Float16* qrow =
      qbuf + ((long)(b * S_ + n * W_ + i0 + nl)) * HD_ + h * 64;
  F16x16 a0, a1;
  a0.q[0] = *(const uint4*)(qrow +      8 * h2);
  a0.q[1] = *(const uint4*)(qrow + 16 + 8 * h2);
  a1.q[0] = *(const uint4*)(qrow + 32 + 8 * h2);
  a1.q[1] = *(const uint4*)(qrow + 48 + 8 * h2);

  // k2 = concat(prev-window, this-window): contiguous rows j in [0, 2W)
  const _Float16* k2 = kbuf + ((long)(b * (NW_ + 1) + n)) * (W_ * HD_);

  // ----- scores: S[m][j] = sum_d Q[m][d] K2[j][d], bias + band mask -----
  for (int jt = 0; jt < 33; ++jt) {
    const int j = i0 + jt * 16 + nl;             // B col n = nl -> K2 row j
    const _Float16* kp = k2 + (long)j * HD_ + h * 64 + 8 * h2;
    F16x16 b0, b1;
    b0.q[0] = *(const uint4*)(kp);       b0.q[1] = *(const uint4*)(kp + 16);
    b1.q[0] = *(const uint4*)(kp + 32);  b1.q[1] = *(const uint4*)(kp + 48);
    v8f acc = {};
    acc = wmma16(a0.v, b0.v, acc);
    acc = wmma16(a1.v, b1.v, acc);
#pragma unroll
    for (int r = 0; r < 8; ++r) {
      const int rowm = 8 * h2 + r;
      const int idx = 512 + rowm - jt * 16 - nl;  // (i0+rowm) + W - j
      const float s =
          (idx >= 0 && idx < 512) ? acc[r] + tb[idx] : -1e30f;
      scw[rowm * SCW + jt * 16 + nl] = s;
    }
  }

  // ----- softmax (unnormalized): row = nl, column half = h2 -----
  const float* srp = scw + nl * SCW + h2 * 272;
  float rmax = -1e30f;
  for (int c = 0; c < 272; c += 4) {
    float4 v = *(const float4*)(srp + c);
    rmax = fmaxf(rmax, fmaxf(fmaxf(v.x, v.y), fmaxf(v.z, v.w)));
  }
  rmax = fmaxf(rmax, __shfl_xor(rmax, 16));
  float ssum = 0.0f;
  _Float16* prp = Pw + nl * SCW + h2 * 272;
  for (int c = 0; c < 272; c += 4) {
    float4 v = *(const float4*)(srp + c);
    float e0 = __expf(v.x - rmax), e1 = __expf(v.y - rmax);
    float e2 = __expf(v.z - rmax), e3 = __expf(v.w - rmax);
    prp[c]     = (_Float16)e0; prp[c + 1] = (_Float16)e1;
    prp[c + 2] = (_Float16)e2; prp[c + 3] = (_Float16)e3;
    ssum += (e0 + e1) + (e2 + e3);
  }
  ssum += __shfl_xor(ssum, 16);   // lane r (and r+16) now holds row r's sum

  // ----- O = P @ V2 (A frags from LDS, B frags contiguous from vT) -----
  const long chunkstride = (long)H_ * D_ * W_;
  const _Float16* vbase =
      vT + ((long)(b * (NW_ + 1) + n) * H_ + h) * (long)(D_ * W_);
  v8f o[4];
  o[0] = {}; o[1] = {}; o[2] = {}; o[3] = {};
  for (int t = 0; t < 17; ++t) {
    const int ck = t * 32;
    F16x16 ap;
    ap.q[0] = *(const uint4*)(Pw + nl * SCW + ck + 16 * h2);
    ap.q[1] = *(const uint4*)(Pw + nl * SCW + ck + 16 * h2 + 8);
    int jg = i0 + ck + 16 * h2;        // global memory row for this K group
    if (jg > 1008) jg = 1008;          // clamp: P is 0 there anyway
    const _Float16* vchunk = vbase + (long)(jg >> 9) * chunkstride + (jg & 511);
#pragma unroll
    for (int g = 0; g < 4; ++g) {
      const _Float16* vp = vchunk + (long)(16 * g + nl) * W_;
      F16x16 bv;
      bv.q[0] = *(const uint4*)(vp);
      bv.q[1] = *(const uint4*)(vp + 8);
      o[g] = wmma16(ap.v, bv.v, o[g]);
    }
  }

  // ----- normalize and write attn_ys (B,S,H,D) -----
#pragma unroll
  for (int r = 0; r < 8; ++r) {
    const int rowm = 8 * h2 + r;
    const float inv = 1.0f / __shfl(ssum, rowm);
    const long ro =
        ((long)(b * S_ + n * W_ + i0 + rowm)) * HD_ + h * 64 + nl;
#pragma unroll
    for (int g = 0; g < 4; ++g)
      out_attn[ro + 16 * g] = o[g][r] * inv;
  }
}

// ---------------------------------------------------------------------------
extern "C" void kernel_launch(void* const* d_in, const int* in_sizes, int n_in,
                              void* d_out, int out_size, void* d_ws, size_t ws_size,
                              hipStream_t stream) {
  (void)in_sizes; (void)n_in; (void)out_size; (void)ws_size;
  const float* xs       = (const float*)d_in[0];
  const float* prev_k   = (const float*)d_in[1];
  const float* prev_v   = (const float*)d_in[2];
  const float* w_q      = (const float*)d_in[3];
  const float* w_k      = (const float*)d_in[4];
  const float* w_v      = (const float*)d_in[5];
  const float* rel_bias = (const float*)d_in[6];

  char* w = (char*)d_ws;
  _Float16* xs_h = (_Float16*)w; w += (size_t)B_ * S_ * E_ * 2;
  _Float16* wt   = (_Float16*)w; w += (size_t)3 * E_ * HD_ * 2;
  _Float16* qbuf = (_Float16*)w; w += (size_t)B_ * S_ * HD_ * 2;
  _Float16* kbuf = (_Float16*)w; w += (size_t)B_ * (NW_ + 1) * W_ * HD_ * 2;
  _Float16* vT   = (_Float16*)w; w += (size_t)B_ * (NW_ + 1) * H_ * D_ * W_ * 2;

  float* out_attn = (float*)d_out;
  float* next_k   = out_attn + (size_t)B_ * S_ * H_ * D_;
  float* next_v   = next_k + (size_t)B_ * W_ * H_ * D_;

  const long convN = (long)B_ * S_ * E_ + 3L * E_ * HD_ + 2L * B_ * W_ * HD_;
  convert_kernel<<<(int)((convN + 255) / 256), 256, 0, stream>>>(
      xs, w_q, w_k, w_v, prev_k, prev_v, xs_h, wt, kbuf, vT);

  proj_kernel<<<dim3((B_ * S_) / 16, 3), 256, 0, stream>>>(
      xs_h, wt, qbuf, kbuf, vT, next_k, next_v);

  const size_t shmem = (size_t)4 * 16 * SCW * 4 + 512 * 4 + (size_t)4 * 16 * SCW * 2;
  attn_kernel<<<dim3(W_ / 64, B_ * NW_ * H_), 128, shmem, stream>>>(
      qbuf, kbuf, vT, rel_bias, out_attn);
}
